// SelfAttention_28071906247306
// MI455X (gfx1250) — matrix operand order repaired and checked by reading, not compiled, not run on hardware
//
#include <hip/hip_runtime.h>

// ---------------------------------------------------------------------------
// CDNA5 / gfx1250 multi-head self-attention (fp32 in/out, bf16 WMMA compute)
// wave32, 256-thread blocks (8 waves). Pipeline:
//   proj_gemm x3 -> scores (WMMA, async global->LDS staging)
//   -> pv_softmax (fused row softmax + attn@V WMMA) -> out_gemm
// ---------------------------------------------------------------------------

typedef __bf16 bf16_t;
typedef __attribute__((ext_vector_type(16))) __bf16 v16bf;
typedef __attribute__((ext_vector_type(8)))  __bf16 v8bf;
typedef __attribute__((ext_vector_type(8)))  float  v8f;
typedef __attribute__((address_space(3)))    __bf16* lds_bf16p;

#define D_MODEL 1024
#define NHEADS  16
#define DKDIM   64
#define SEQ     2048
#define BATCH   2
#define MTOT    (BATCH * SEQ)       // 4096 rows for the dense GEMMs
#define NEGVAL  (-1e9f)

static __device__ __forceinline__ v16bf concat8(v8bf lo, v8bf hi) {
  return __builtin_shufflevector(lo, hi, 0,1,2,3,4,5,6,7,8,9,10,11,12,13,14,15);
}

static __device__ __forceinline__ v8f wmma_bf16(v16bf a, v16bf b, v8f c) {
  // (neg_a, A, neg_b, B, c_mod, C, reuse_a, reuse_b)
  return __builtin_amdgcn_wmma_f32_16x16x32_bf16(false, a, false, b, (short)0, c,
                                                 false, false);
}

// Load A fragment (16x32 bf16) from row-major LDS tile with row stride `ld`.
// ISA layout: lanes<16 hold K = {0..7, 16..23}, lanes>=16 hold K = {8..15, 24..31}.
static __device__ __forceinline__ v16bf load_a_frag(const bf16_t* tile, int ld,
                                                    int row, int k0, int lane) {
  const bf16_t* r = tile + row * ld + k0 + ((lane >> 4) << 3);
  v8bf lo = *(const v8bf*)(r);
  v8bf hi = *(const v8bf*)(r + 16);
  return concat8(lo, hi);
}

// Load B fragment (32x16 bf16) from N-major LDS tile ([n][k], row stride `ld`).
// ISA layout: lanes<16 hold K = {0..7, 16..23}, lanes>=16 hold K = {8..15, 24..31}.
static __device__ __forceinline__ v16bf load_b_frag(const bf16_t* tile, int ld,
                                                    int col, int k0, int lane) {
  const bf16_t* r = tile + col * ld + k0 + ((lane >> 4) << 3);
  v8bf lo = *(const v8bf*)(r);
  v8bf hi = *(const v8bf*)(r + 16);
  return concat8(lo, hi);
}

static __device__ __forceinline__ void cvt_store8(bf16_t* dst, float4 a, float4 b) {
  v8bf p;
  p[0] = (bf16_t)a.x; p[1] = (bf16_t)a.y; p[2] = (bf16_t)a.z; p[3] = (bf16_t)a.w;
  p[4] = (bf16_t)b.x; p[5] = (bf16_t)b.y; p[6] = (bf16_t)b.z; p[7] = (bf16_t)b.w;
  *(v8bf*)dst = p;
}

// ---------------------------------------------------------------------------
// Kernel 1: QKV projection.  Out[b,h,s,dk] (bf16) = (X @ W + bias), head-split.
// Block tile 128(M) x 64(N), K-step 32.  8 waves -> each wave 32x32 (2x2 frags).
// ---------------------------------------------------------------------------
__global__ __launch_bounds__(256) void proj_gemm(
    const float* __restrict__ X,      // [4096,1024] fp32
    const float* __restrict__ W,      // [1024,1024] fp32
    const float* __restrict__ bias,   // [1024]
    bf16_t* __restrict__ Out)         // [B,H,S,64] bf16
{
  __shared__ bf16_t As[128][32];      // 8 KB
  __shared__ bf16_t Bs[64][32];       // 4 KB
  const int tid   = threadIdx.x;
  const int lane  = tid & 31;
  const int wave  = tid >> 5;
  const int waveM = wave >> 1;        // 0..3
  const int waveN = wave & 1;         // 0..1
  const int mBase = blockIdx.y * 128;
  const int nBase = blockIdx.x * 64;

  v8f acc[2][2] = {};

  const int arow = tid >> 1;          // staging: A row per thread
  const int acol = (tid & 1) << 4;    // 0 or 16

  for (int k0 = 0; k0 < D_MODEL; k0 += 32) {
    // Stage A (fp32 -> bf16): 128x32 tile, 16 elems/thread.
    {
      const float* src = X + (size_t)(mBase + arow) * D_MODEL + k0 + acol;
      float4 f0 = ((const float4*)src)[0];
      float4 f1 = ((const float4*)src)[1];
      float4 f2 = ((const float4*)src)[2];
      float4 f3 = ((const float4*)src)[3];
      cvt_store8(&As[arow][acol],     f0, f1);
      cvt_store8(&As[arow][acol + 8], f2, f3);
      if (k0 + 32 < D_MODEL) __builtin_prefetch(src + 32, 0, 1);
    }
    // Stage B (fp32 -> bf16, transpose to [n][k]): 32x64 tile, 8 elems/thread.
#pragma unroll
    for (int i = 0; i < 8; ++i) {
      int e  = i * 256 + tid;         // 0..2047
      int kk = e >> 6;                // 0..31
      int nn = e & 63;
      Bs[nn][kk] = (bf16_t)W[(size_t)(k0 + kk) * D_MODEL + nBase + nn];
    }
    __syncthreads();

    v16bf af[2], bfr[2];
#pragma unroll
    for (int fm = 0; fm < 2; ++fm)
      af[fm] = load_a_frag(&As[0][0], 32, waveM * 32 + fm * 16 + (lane & 15), 0, lane);
#pragma unroll
    for (int fn = 0; fn < 2; ++fn)
      bfr[fn] = load_b_frag(&Bs[0][0], 32, waveN * 32 + fn * 16 + (lane & 15), 0, lane);
#pragma unroll
    for (int fm = 0; fm < 2; ++fm)
#pragma unroll
      for (int fn = 0; fn < 2; ++fn)
        acc[fm][fn] = wmma_bf16(af[fm], bfr[fn], acc[fm][fn]);
    __syncthreads();
  }

  // Epilogue: bias add, convert to bf16, head-split store [B,H,S,64].
#pragma unroll
  for (int fn = 0; fn < 2; ++fn) {
    const int col = nBase + waveN * 32 + fn * 16 + (lane & 15);
    const int h   = col >> 6;
    const int dk  = col & 63;
    const float bc = bias[col];
#pragma unroll
    for (int fm = 0; fm < 2; ++fm) {
#pragma unroll
      for (int i = 0; i < 8; ++i) {
        int gm = mBase + waveM * 32 + fm * 16 + i + ((lane >> 4) << 3);
        int b  = gm >> 11;            // /SEQ
        int s  = gm & (SEQ - 1);
        Out[(((size_t)(b * NHEADS + h)) * SEQ + s) * DKDIM + dk] =
            (bf16_t)(acc[fm][fn][i] + bc);
      }
    }
  }
}

// ---------------------------------------------------------------------------
// Kernel 2: raw scores = (Q @ K^T) / 8, masked (causal | pad) -> fp32 attn.
// Q/K tiles staged with GLOBAL_LOAD_ASYNC_TO_LDS_B128 (ASYNCcnt path).
// ---------------------------------------------------------------------------
__global__ __launch_bounds__(256) void scores_kernel(
    const bf16_t* __restrict__ Q,     // [BH,S,64]
    const bf16_t* __restrict__ Km,    // [BH,S,64]
    const int* __restrict__ token_ids,// [B,S]
    float* __restrict__ attn)         // [BH,S,S]
{
  const int tid   = threadIdx.x;
  const int bh    = blockIdx.z;
  const int b     = bh >> 4;
  const int mBase = blockIdx.y * 128; // query rows
  const int nBase = blockIdx.x * 128; // key cols

  // Fast path: tile entirely above the causal diagonal -> pure -1e9 fill.
  if (nBase >= mBase + 128) {
#pragma unroll 4
    for (int i = 0; i < 64; ++i) {
      int e = i * 256 + tid;          // 0..16383
      int r = e >> 7, c = e & 127;
      attn[((size_t)(bh * SEQ + mBase + r)) * SEQ + nBase + c] = NEGVAL;
    }
    return;
  }

  __shared__ bf16_t Qs[128][64];      // 16 KB
  __shared__ bf16_t Ks[128][64];      // 16 KB

  // Stage whole Q and K tiles via async global->LDS copies (16B per lane).
  const bf16_t* Qg = Q + (size_t)bh * SEQ * DKDIM;
  const bf16_t* Kg = Km + (size_t)bh * SEQ * DKDIM;
#pragma unroll
  for (int i = 0; i < 4; ++i) {
    int ch  = i * 256 + tid;          // 0..1023 16B chunks
    int row = ch >> 3;
    int cb  = (ch & 7) << 3;          // bf16 element offset
    unsigned qdst = (unsigned)(unsigned long)(lds_bf16p)&Qs[row][cb];
    unsigned kdst = (unsigned)(unsigned long)(lds_bf16p)&Ks[row][cb];
    const bf16_t* qsrc = Qg + (size_t)(mBase + row) * DKDIM + cb;
    const bf16_t* ksrc = Kg + (size_t)(nBase + row) * DKDIM + cb;
    asm volatile("global_load_async_to_lds_b128 %0, %1, off"
                 :: "v"(qdst), "v"(qsrc) : "memory");
    asm volatile("global_load_async_to_lds_b128 %0, %1, off"
                 :: "v"(kdst), "v"(ksrc) : "memory");
  }
  asm volatile("s_wait_asynccnt 0x0" ::: "memory");
  __syncthreads();

  const int lane  = tid & 31;
  const int wave  = tid >> 5;
  const int waveM = wave >> 1;        // 0..3 (32 q-rows each)
  const int waveN = wave & 1;         // 0..1 (64 k-cols each)

  v8f acc[2][4] = {};
#pragma unroll
  for (int ks = 0; ks < 2; ++ks) {    // dk = 64 -> two K-steps of 32
    const int k0 = ks * 32;
    v16bf af[2], bfr[4];
#pragma unroll
    for (int fm = 0; fm < 2; ++fm)
      af[fm] = load_a_frag(&Qs[0][0], 64, waveM * 32 + fm * 16 + (lane & 15), k0, lane);
#pragma unroll
    for (int fn = 0; fn < 4; ++fn)
      bfr[fn] = load_b_frag(&Ks[0][0], 64, waveN * 64 + fn * 16 + (lane & 15), k0, lane);
#pragma unroll
    for (int fm = 0; fm < 2; ++fm)
#pragma unroll
      for (int fn = 0; fn < 4; ++fn)
        acc[fm][fn] = wmma_bf16(af[fm], bfr[fn], acc[fm][fn]);
  }

  const float scale = 0.125f;         // 1/sqrt(64)
#pragma unroll
  for (int fn = 0; fn < 4; ++fn) {
    const int gk  = nBase + waveN * 64 + fn * 16 + (lane & 15);
    const bool padk = (token_ids[b * SEQ + gk] == 0);
#pragma unroll
    for (int fm = 0; fm < 2; ++fm) {
#pragma unroll
      for (int i = 0; i < 8; ++i) {
        int gq = mBase + waveM * 32 + fm * 16 + i + ((lane >> 4) << 3);
        float v = acc[fm][fn][i] * scale;
        bool masked = padk || (gk > gq);
        attn[((size_t)(bh * SEQ + gq)) * SEQ + gk] = masked ? NEGVAL : v;
      }
    }
  }
}

// ---------------------------------------------------------------------------
// Kernel 3: fused softmax + PV.  For a 128-row strip of one head:
//   sweep 1: per-row online max / sum-exp over raw scores (strip stays in L2)
//   fill   : cols beyond causal edge -> 0 (or 1/S for all-masked rows)
//   sweep 2: p = exp(s-m)/l, written back to attn AND fed (bf16) into WMMA
//            accumulation of attn @ V.  K-loop truncated at the causal edge.
// Output merged-heads bf16 [B,S,1024].
// ---------------------------------------------------------------------------
__global__ __launch_bounds__(256) void pv_softmax_kernel(
    float* __restrict__ attn,         // [BH,S,S] raw in, probs out
    const bf16_t* __restrict__ V,     // [BH,S,64]
    bf16_t* __restrict__ AO)          // [B,S,1024]
{
  __shared__ bf16_t As[128][32];
  __shared__ bf16_t Bs[64][32];
  __shared__ float  Mh[128][2], Lh[128][2];
  __shared__ float  Mrow[128], InvL[128];

  const int tid   = threadIdx.x;
  const int lane  = tid & 31;
  const int wave  = tid >> 5;
  const int waveM = wave >> 1;
  const int waveN = wave & 1;
  const int bh    = blockIdx.y;
  const int b     = bh >> 4;
  const int h     = bh & 15;
  const int mBase = blockIdx.x * 128;
  const int kEnd  = mBase + 128;      // causal edge (multiple of 128)

  float* Ag = attn + (size_t)bh * SEQ * SEQ;
  const bf16_t* Vg = V + (size_t)bh * SEQ * DKDIM;

  // ---- Sweep 1: per-row running max & sum of exp (2 threads per row).
  const int r    = tid >> 1;
  const int half = tid & 1;
  {
    const int c0 = half * (kEnd >> 1);
    const int c1 = c0 + (kEnd >> 1);
    const float* rowp = Ag + (size_t)(mBase + r) * SEQ;
    float m = -3.0e38f, l = 0.f;
    for (int c = c0; c < c1; c += 4) {
      float4 f = *(const float4*)(rowp + c);
      float m4 = fmaxf(fmaxf(f.x, f.y), fmaxf(f.z, f.w));
      if (m4 > m) { l *= __expf(m - m4); m = m4; }
      l += __expf(f.x - m) + __expf(f.y - m) +
           __expf(f.z - m) + __expf(f.w - m);
    }
    Mh[r][half] = m;
    Lh[r][half] = l;
  }
  __syncthreads();
  if (tid < 128) {
    float m0 = Mh[tid][0], m1 = Mh[tid][1];
    float mm = fmaxf(m0, m1);
    float ll = Lh[tid][0] * __expf(m0 - mm) + Lh[tid][1] * __expf(m1 - mm);
    if (mm <= -1.0e8f) ll = (float)SEQ;  // all-masked row -> uniform 1/S
    Mrow[tid] = mm;
    InvL[tid] = 1.f / ll;
  }
  __syncthreads();

  // ---- Fill cols >= causal edge of this strip with final probabilities.
  {
    const int width = SEQ - kEnd;
    if (width > 0) {
      float fv = (Mrow[r] <= -1.0e8f) ? InvL[r] : 0.f;
      float4 f4 = make_float4(fv, fv, fv, fv);
      float* rowq = Ag + (size_t)(mBase + r) * SEQ + kEnd + half * (width >> 1);
      for (int c = 0; c < (width >> 1); c += 4) ((float4*)rowq)[c >> 2] = f4;
    }
  }

  // ---- Sweep 2: normalize + write back + WMMA accumulate attn @ V.
  v8f acc[2][2] = {};
  const int arow = tid >> 1;
  const int acol = (tid & 1) << 4;
  const float mr = Mrow[arow];
  const float il = InvL[arow];

  for (int k0 = 0; k0 < kEnd; k0 += 32) {
    {
      float* p = Ag + (size_t)(mBase + arow) * SEQ + k0 + acol;
      float4 f0 = ((const float4*)p)[0];
      float4 f1 = ((const float4*)p)[1];
      float4 f2 = ((const float4*)p)[2];
      float4 f3 = ((const float4*)p)[3];
      f0.x = __expf(f0.x - mr) * il; f0.y = __expf(f0.y - mr) * il;
      f0.z = __expf(f0.z - mr) * il; f0.w = __expf(f0.w - mr) * il;
      f1.x = __expf(f1.x - mr) * il; f1.y = __expf(f1.y - mr) * il;
      f1.z = __expf(f1.z - mr) * il; f1.w = __expf(f1.w - mr) * il;
      f2.x = __expf(f2.x - mr) * il; f2.y = __expf(f2.y - mr) * il;
      f2.z = __expf(f2.z - mr) * il; f2.w = __expf(f2.w - mr) * il;
      f3.x = __expf(f3.x - mr) * il; f3.y = __expf(f3.y - mr) * il;
      f3.z = __expf(f3.z - mr) * il; f3.w = __expf(f3.w - mr) * il;
      ((float4*)p)[0] = f0; ((float4*)p)[1] = f1;
      ((float4*)p)[2] = f2; ((float4*)p)[3] = f3;
      cvt_store8(&As[arow][acol],     f0, f1);
      cvt_store8(&As[arow][acol + 8], f2, f3);
      if (k0 + 32 < kEnd) __builtin_prefetch(p + 32, 0, 1);
    }
#pragma unroll
    for (int i = 0; i < 8; ++i) {
      int e  = i * 256 + tid;
      int kk = e >> 6;
      int nn = e & 63;
      Bs[nn][kk] = Vg[(size_t)(k0 + kk) * DKDIM + nn];
    }
    __syncthreads();

    v16bf af[2], bfr[2];
#pragma unroll
    for (int fm = 0; fm < 2; ++fm)
      af[fm] = load_a_frag(&As[0][0], 32, waveM * 32 + fm * 16 + (lane & 15), 0, lane);
#pragma unroll
    for (int fn = 0; fn < 2; ++fn)
      bfr[fn] = load_b_frag(&Bs[0][0], 32, waveN * 32 + fn * 16 + (lane & 15), 0, lane);
#pragma unroll
    for (int fm = 0; fm < 2; ++fm)
#pragma unroll
      for (int fn = 0; fn < 2; ++fn)
        acc[fm][fn] = wmma_bf16(af[fm], bfr[fn], acc[fm][fn]);
    __syncthreads();
  }

#pragma unroll
  for (int fn = 0; fn < 2; ++fn) {
    const int col = waveN * 32 + fn * 16 + (lane & 15);  // 0..63 (dk)
#pragma unroll
    for (int fm = 0; fm < 2; ++fm) {
#pragma unroll
      for (int i = 0; i < 8; ++i) {
        int gm = mBase + waveM * 32 + fm * 16 + i + ((lane >> 4) << 3);
        AO[((size_t)(b * SEQ + gm)) * D_MODEL + h * DKDIM + col] =
            (bf16_t)acc[fm][fn][i];
      }
    }
  }
}

// ---------------------------------------------------------------------------
// Kernel 4: final projection.  out (fp32) = AO(bf16) @ wo + bo.
// ---------------------------------------------------------------------------
__global__ __launch_bounds__(256) void out_gemm(
    const bf16_t* __restrict__ A,     // [4096,1024] bf16
    const float* __restrict__ W,      // [1024,1024] fp32
    const float* __restrict__ bias,   // [1024]
    float* __restrict__ Out)          // [4096,1024] fp32
{
  __shared__ bf16_t As[128][32];
  __shared__ bf16_t Bs[64][32];
  const int tid   = threadIdx.x;
  const int lane  = tid & 31;
  const int wave  = tid >> 5;
  const int waveM = wave >> 1;
  const int waveN = wave & 1;
  const int mBase = blockIdx.y * 128;
  const int nBase = blockIdx.x * 64;

  v8f acc[2][2] = {};
  const int arow = tid >> 1;
  const int acol = (tid & 1) << 4;

  for (int k0 = 0; k0 < D_MODEL; k0 += 32) {
    {
      const bf16_t* src = A + (size_t)(mBase + arow) * D_MODEL + k0 + acol;
      *(uint4*)&As[arow][acol]     = ((const uint4*)src)[0];
      *(uint4*)&As[arow][acol + 8] = ((const uint4*)src)[1];
      if (k0 + 32 < D_MODEL) __builtin_prefetch(src + 32, 0, 1);
    }
#pragma unroll
    for (int i = 0; i < 8; ++i) {
      int e  = i * 256 + tid;
      int kk = e >> 6;
      int nn = e & 63;
      Bs[nn][kk] = (bf16_t)W[(size_t)(k0 + kk) * D_MODEL + nBase + nn];
    }
    __syncthreads();

    v16bf af[2], bfr[2];
#pragma unroll
    for (int fm = 0; fm < 2; ++fm)
      af[fm] = load_a_frag(&As[0][0], 32, waveM * 32 + fm * 16 + (lane & 15), 0, lane);
#pragma unroll
    for (int fn = 0; fn < 2; ++fn)
      bfr[fn] = load_b_frag(&Bs[0][0], 32, waveN * 32 + fn * 16 + (lane & 15), 0, lane);
#pragma unroll
    for (int fm = 0; fm < 2; ++fm)
#pragma unroll
      for (int fn = 0; fn < 2; ++fn)
        acc[fm][fn] = wmma_bf16(af[fm], bfr[fn], acc[fm][fn]);
    __syncthreads();
  }

#pragma unroll
  for (int fn = 0; fn < 2; ++fn) {
    const int col = nBase + waveN * 32 + fn * 16 + (lane & 15);
    const float bc = bias[col];
#pragma unroll
    for (int fm = 0; fm < 2; ++fm) {
#pragma unroll
      for (int i = 0; i < 8; ++i) {
        int gm = mBase + waveM * 32 + fm * 16 + i + ((lane >> 4) << 3);
        Out[(size_t)gm * D_MODEL + col] = acc[fm][fn][i] + bc;
      }
    }
  }
}

// ---------------------------------------------------------------------------
// Host launcher
// ---------------------------------------------------------------------------
extern "C" void kernel_launch(void* const* d_in, const int* in_sizes, int n_in,
                              void* d_out, int out_size, void* d_ws, size_t ws_size,
                              hipStream_t stream) {
  const float* x   = (const float*)d_in[0];
  const int*   tok = (const int*)d_in[1];
  const float* wq  = (const float*)d_in[2];
  const float* bq  = (const float*)d_in[3];
  const float* wk  = (const float*)d_in[4];
  const float* bk  = (const float*)d_in[5];
  const float* wv  = (const float*)d_in[6];
  const float* bv  = (const float*)d_in[7];
  const float* wo  = (const float*)d_in[8];
  const float* bo  = (const float*)d_in[9];

  float* out  = (float*)d_out;                       // [B,S,1024]  (4,194,304)
  float* attn = out + (size_t)MTOT * D_MODEL;        // [B,H,S,S]   (134,217,728)

  // Workspace: Q,K,V head-split bf16 + merged-heads PV output bf16 (32 MB).
  bf16_t* Q  = (bf16_t*)d_ws;
  bf16_t* K  = Q + (size_t)MTOT * D_MODEL;
  bf16_t* V  = K + (size_t)MTOT * D_MODEL;
  bf16_t* AO = V + (size_t)MTOT * D_MODEL;

  dim3 blk(256);
  proj_gemm<<<dim3(D_MODEL / 64, MTOT / 128), blk, 0, stream>>>(x, wq, bq, Q);
  proj_gemm<<<dim3(D_MODEL / 64, MTOT / 128), blk, 0, stream>>>(x, wk, bk, K);
  proj_gemm<<<dim3(D_MODEL / 64, MTOT / 128), blk, 0, stream>>>(x, wv, bv, V);

  scores_kernel<<<dim3(SEQ / 128, SEQ / 128, BATCH * NHEADS), blk, 0, stream>>>(
      Q, K, tok, attn);

  pv_softmax_kernel<<<dim3(SEQ / 128, BATCH * NHEADS), blk, 0, stream>>>(
      attn, V, AO);

  out_gemm<<<dim3(D_MODEL / 64, MTOT / 128), blk, 0, stream>>>(AO, wo, bo, out);
}
